// MyScaledDotProductWithBoxAttentionTwo_65566970741301
// MI455X (gfx1250) — compile-verified, matrix operand order checked by reading, not run on hardware
//
#include <hip/hip_runtime.h>
#include <hip/hip_bf16.h>

typedef _Float16 h16;
typedef __attribute__((ext_vector_type(16))) _Float16 v16h;
typedef __attribute__((ext_vector_type(8)))  _Float16 v8h;
typedef __attribute__((ext_vector_type(8)))  float    v8f;

#define HH 8
#define DHEAD 64
#define NSEQ 2048
#define DM 512
#define BSZ 2
#define MROWS (BSZ * NSEQ)          // 4096

// WMMA 16-bit fragment mapping (A operand 16x32, B operand 32x16):
//   lane = (row_or_col & 15) + 16*((k>>3)&1)
//   slot = (k&7) + 8*((k>>4)&1)
// => for an aligned run of 8 consecutive k: lane constant, slots contiguous.
__device__ __forceinline__ int frag_lane(int rc, int k) { return (rc & 15) + 16 * ((k >> 3) & 1); }
__device__ __forceinline__ int frag_idx(int k)          { return (k & 7) + 8 * ((k >> 4) & 1); }

__device__ __forceinline__ size_t heads_off(int gm, int c) {
    int b = gm >> 11, n = gm & (NSEQ - 1);      // NSEQ = 2048
    int h = c >> 6, d = c & 63;
    return (((size_t)b * HH + h) * NSEQ + n) * DHEAD + d;
}

// ---------------------------------------------------------------------------
// NT GEMM: C[M,512] = scale * (A[M,512] @ W[512,512]^T + bias (+ Z))
// Tile 128x64, K-chunks of 32; 8 waves x (16 rows x 64 cols).
// ---------------------------------------------------------------------------
template <bool A_HEADS, bool C_F32, bool HAS_Z>
__global__ __launch_bounds__(256)
void gemm_nt_kernel(const float* __restrict__ Af32, const h16* __restrict__ Ah16,
                    const float* __restrict__ W, const float* __restrict__ bias,
                    const h16* __restrict__ Zadd, float scale,
                    float* __restrict__ Cf32, h16* __restrict__ Ch16)
{
    __shared__ __align__(32) h16 ldsA[8 * 32 * 16];
    __shared__ __align__(32) h16 ldsB[4 * 32 * 16];

    const int tid  = threadIdx.x;
    const int lane = tid & 31;
    const int wave = tid >> 5;
    const int n0 = blockIdx.x * 64;
    const int m0 = blockIdx.y * 128;

    v8f acc[4] = {};

    for (int k0 = 0; k0 < DM; k0 += 32) {
        // prefetch next A/B tiles (global_prefetch_b8)
        if (k0 + 32 < DM) {
            if constexpr (!A_HEADS)
                __builtin_prefetch(&Af32[(size_t)(m0 + (tid >> 1)) * DM + (k0 + 32) + (tid & 1) * 16], 0, 3);
            __builtin_prefetch(&W[(size_t)(n0 + (tid >> 2)) * DM + (k0 + 32) + (tid & 3) * 8], 0, 3);
        }
        // stage A tile 128x32: 512 groups of 8k; 2 groups/thread; one v8h store each
        #pragma unroll
        for (int e = 0; e < 2; ++e) {
            int g  = e * 256 + tid;             // 0..511
            int r  = g >> 2;                    // row 0..127
            int kg = (g & 3) * 8;               // k-group base
            int gm = m0 + r, gk = k0 + kg;
            v8h hv;
            if constexpr (A_HEADS) {
                hv = *(const v8h*)&Ah16[heads_off(gm, gk)];
            } else {
                v8f fv = *(const v8f*)&Af32[(size_t)gm * DM + gk];
                hv = __builtin_convertvector(fv, v8h);
            }
            int base = ((r >> 4) * 32 + frag_lane(r & 15, kg)) * 16 + frag_idx(kg);
            *(v8h*)&ldsA[base] = hv;
        }
        // stage B tile 64x32: 256 groups; 1 group/thread
        {
            int g  = tid;
            int c  = g >> 2;
            int kg = (g & 3) * 8;
            v8f fv = *(const v8f*)&W[(size_t)(n0 + c) * DM + (k0 + kg)];
            v8h hv = __builtin_convertvector(fv, v8h);
            int base = ((c >> 4) * 32 + frag_lane(c & 15, kg)) * 16 + frag_idx(kg);
            *(v8h*)&ldsB[base] = hv;
        }
        __syncthreads();

        v16h a = *(const v16h*)&ldsA[(wave * 32 + lane) * 16];
        #pragma unroll
        for (int t = 0; t < 4; ++t) {
            v16h bb = *(const v16h*)&ldsB[(t * 32 + lane) * 16];
            acc[t] = __builtin_amdgcn_wmma_f32_16x16x32_f16(false, a, false, bb,
                                                            (short)0, acc[t], false, false);
        }
        __syncthreads();
    }

    const int rbase = (lane >> 4) * 8;
    #pragma unroll
    for (int t = 0; t < 4; ++t) {
        int c = n0 + t * 16 + (lane & 15);
        float bv = bias[c];
        #pragma unroll
        for (int j = 0; j < 8; ++j) {
            int gm = m0 + wave * 16 + rbase + j;
            float v = acc[t][j] + bv;
            if constexpr (HAS_Z) v += (float)Zadd[heads_off(gm, c)];
            v *= scale;
            if constexpr (C_F32) Cf32[(size_t)gm * DM + c] = v;
            else                 Ch16[heads_off(gm, c)] = (h16)v;
        }
    }
}

// ---------------------------------------------------------------------------
// Flash attention: block = (b, h, 128-query tile); 8 waves x 16 queries.
// 64-key blocks; scores stay in registers/LDS. Q pre-scaled by 1/sqrt(dk).
// ---------------------------------------------------------------------------
__global__ __launch_bounds__(256)
void flash_attn_kernel(const h16* __restrict__ Q, const h16* __restrict__ Kt,
                       const h16* __restrict__ Vt, const int* __restrict__ mask,
                       const float* __restrict__ wts, h16* __restrict__ O)
{
    __shared__ __align__(32) h16 ldsQ[8 * 2 * 32 * 16];
    __shared__ __align__(32) h16 ldsK[4 * 2 * 32 * 16];
    __shared__ __align__(32) h16 ldsV[4 * 2 * 32 * 16];
    __shared__ __align__(32) h16 ldsP[8 * 2 * 32 * 16];

    const int tid  = threadIdx.x;
    const int lane = tid & 31;
    const int wave = tid >> 5;
    const int b = blockIdx.z, h = blockIdx.y;
    const int q0 = blockIdx.x * 128;

    const h16* Qbh = Q  + ((size_t)b * HH + h) * NSEQ * DHEAD;
    const h16* Kbh = Kt + ((size_t)b * HH + h) * NSEQ * DHEAD;
    const h16* Vbh = Vt + ((size_t)b * HH + h) * NSEQ * DHEAD;
    const int*   mk = mask + (size_t)b * NSEQ;
    const float* w  = wts  + (size_t)b * NSEQ;

    // stage Q tile 128x64: 1024 groups of 8; 4/thread; vector loads + vector LDS stores
    #pragma unroll
    for (int e = 0; e < 4; ++e) {
        int g  = e * 256 + tid;                // 0..1023
        int r  = g >> 3;                       // row 0..127
        int dg = (g & 7) * 8;                  // d-group base
        v8h hv = *(const v8h*)&Qbh[(size_t)(q0 + r) * DHEAD + dg];
        int ch = dg >> 5, kk = dg & 31;
        int base = (((r >> 4) * 2 + ch) * 32 + frag_lane(r & 15, kk)) * 16 + frag_idx(kk);
        *(v8h*)&ldsQ[base] = hv;
    }
    __syncthreads();
    v16h aq0 = *(const v16h*)&ldsQ[((wave * 2 + 0) * 32 + lane) * 16];
    v16h aq1 = *(const v16h*)&ldsQ[((wave * 2 + 1) * 32 + lane) * 16];

    float mrow[8], lrow[8];
    #pragma unroll
    for (int j = 0; j < 8; ++j) { mrow[j] = -__builtin_inff(); lrow[j] = 0.0f; }
    v8f acc[4] = {};
    const int hlf = lane >> 4;

    for (int k0 = 0; k0 < NSEQ; k0 += 64) {
        // prefetch next key block (64x64 f16 = 8KB each => 16 halfs/thread)
        if (k0 + 64 < NSEQ) {
            __builtin_prefetch(&Kbh[(size_t)(k0 + 64) * DHEAD + tid * 16], 0, 3);
            __builtin_prefetch(&Vbh[(size_t)(k0 + 64) * DHEAD + tid * 16], 0, 3);
        }
        // stage K/V 64x64: 512 groups of 8; 2/thread
        #pragma unroll
        for (int e = 0; e < 2; ++e) {
            int g  = e * 256 + tid;            // 0..511
            int r  = g >> 3;                   // key 0..63
            int dg = (g & 7) * 8;
            // K as B-operand of QK^T (contraction = d): vector store
            v8h kv = *(const v8h*)&Kbh[(size_t)(k0 + r) * DHEAD + dg];
            int kch = dg >> 5, kkk = dg & 31;
            int kbase = (((r >> 4) * 2 + kch) * 32 + frag_lane(r & 15, kkk)) * 16 + frag_idx(kkk);
            *(v8h*)&ldsK[kbase] = kv;
            // V as B-operand of PV (contraction = key): lane varies with d -> 8 b16 stores
            v8h vv = *(const v8h*)&Vbh[(size_t)(k0 + r) * DHEAD + dg];
            int vkch = r >> 5, vkk = r & 31;
            int vbase = (((dg >> 4) * 2 + vkch) * 32 + frag_lane(dg & 15, vkk)) * 16 + frag_idx(vkk);
            #pragma unroll
            for (int i = 0; i < 8; ++i) ldsV[vbase + i * 16] = vv[i];
        }
        __syncthreads();

        // S = Q.K^T
        v8f s[4];
        #pragma unroll
        for (int t = 0; t < 4; ++t) {
            v8f z = {};
            v16h b0 = *(const v16h*)&ldsK[((t * 2 + 0) * 32 + lane) * 16];
            v16h b1 = *(const v16h*)&ldsK[((t * 2 + 1) * 32 + lane) * 16];
            z = __builtin_amdgcn_wmma_f32_16x16x32_f16(false, aq0, false, b0, (short)0, z, false, false);
            z = __builtin_amdgcn_wmma_f32_16x16x32_f16(false, aq1, false, b1, (short)0, z, false, false);
            s[t] = z;
        }

        // per-key weight + mask, running row max
        float newm[8];
        #pragma unroll
        for (int j = 0; j < 8; ++j) newm[j] = mrow[j];
        #pragma unroll
        for (int t = 0; t < 4; ++t) {
            int key = k0 + t * 16 + (lane & 15);
            float wv = w[key];
            bool msk = (mk[key] != 0);
            #pragma unroll
            for (int j = 0; j < 8; ++j) {
                float v = msk ? -__builtin_inff() : s[t][j] * wv;
                s[t][j] = v;
                newm[j] = fmaxf(newm[j], v);
            }
        }
        #pragma unroll
        for (int j = 0; j < 8; ++j) {
            float v = newm[j];
            v = fmaxf(v, __shfl_xor(v, 1, 32));
            v = fmaxf(v, __shfl_xor(v, 2, 32));
            v = fmaxf(v, __shfl_xor(v, 4, 32));
            v = fmaxf(v, __shfl_xor(v, 8, 32));
            newm[j] = v;
        }

        float rsum[8];
        #pragma unroll
        for (int j = 0; j < 8; ++j) rsum[j] = 0.0f;
        #pragma unroll
        for (int t = 0; t < 4; ++t)
            #pragma unroll
            for (int j = 0; j < 8; ++j) {
                float v = s[t][j];
                float p = (v == -__builtin_inff()) ? 0.0f : __expf(v - newm[j]);
                s[t][j] = p;
                rsum[j] += p;
            }
        #pragma unroll
        for (int j = 0; j < 8; ++j) {
            float v = rsum[j];
            v += __shfl_xor(v, 1, 32);
            v += __shfl_xor(v, 2, 32);
            v += __shfl_xor(v, 4, 32);
            v += __shfl_xor(v, 8, 32);
            float sc = (mrow[j] == -__builtin_inff()) ? 0.0f : __expf(mrow[j] - newm[j]);
            lrow[j] = lrow[j] * sc + v;
            mrow[j] = newm[j];
            #pragma unroll
            for (int t = 0; t < 4; ++t) acc[t][j] *= sc;
        }

        // transpose P (C-layout regs) -> per-wave A-fragment LDS region
        h16* Pw = &ldsP[wave * 2 * 32 * 16];
        #pragma unroll
        for (int t = 0; t < 4; ++t) {
            int c = t * 16 + (lane & 15);
            int ch = c >> 5, kk = c & 31;
            int pbase = (ch * 32 + frag_lane(8 * hlf, kk)) * 16 + frag_idx(kk);
            #pragma unroll
            for (int j = 0; j < 8; ++j)
                Pw[pbase + j * 16] = (h16)s[t][j];    // lane' = j + base, stride 16 halfs
        }
        v16h ap0 = *(const v16h*)&Pw[(0 * 32 + lane) * 16];
        v16h ap1 = *(const v16h*)&Pw[(1 * 32 + lane) * 16];

        // O += P.V
        #pragma unroll
        for (int t = 0; t < 4; ++t) {
            v16h b0 = *(const v16h*)&ldsV[((t * 2 + 0) * 32 + lane) * 16];
            v16h b1 = *(const v16h*)&ldsV[((t * 2 + 1) * 32 + lane) * 16];
            acc[t] = __builtin_amdgcn_wmma_f32_16x16x32_f16(false, ap0, false, b0, (short)0, acc[t], false, false);
            acc[t] = __builtin_amdgcn_wmma_f32_16x16x32_f16(false, ap1, false, b1, (short)0, acc[t], false, false);
        }
        __syncthreads();
    }

    #pragma unroll
    for (int j = 0; j < 8; ++j) {
        float inv = (lrow[j] > 0.0f) ? 1.0f / lrow[j] : 0.0f;
        #pragma unroll
        for (int t = 0; t < 4; ++t) acc[t][j] *= inv;
    }
    h16* Obh = O + ((size_t)b * HH + h) * NSEQ * DHEAD;
    #pragma unroll
    for (int t = 0; t < 4; ++t) {
        int d = t * 16 + (lane & 15);
        #pragma unroll
        for (int j = 0; j < 8; ++j) {
            int q = q0 + wave * 16 + j + 8 * hlf;
            Obh[(size_t)q * DHEAD + d] = (h16)acc[t][j];
        }
    }
}

extern "C" void kernel_launch(void* const* d_in, const int* in_sizes, int n_in,
                              void* d_out, int out_size, void* d_ws, size_t ws_size,
                              hipStream_t stream)
{
    const float* regions   = (const float*)d_in[0];
    const float* grids     = (const float*)d_in[1];
    const float* interests = (const float*)d_in[2];
    const int*   mask_r    = (const int*)d_in[3];
    const int*   mask_g    = (const int*)d_in[4];
    const float* attw      = (const float*)d_in[5];
    const float* Wq  = (const float*)d_in[6];  const float* bq   = (const float*)d_in[7];
    const float* Wk  = (const float*)d_in[8];  const float* bk   = (const float*)d_in[9];
    const float* Wv  = (const float*)d_in[10]; const float* bv   = (const float*)d_in[11];
    const float* Wq12= (const float*)d_in[12]; const float* bq12 = (const float*)d_in[13];
    const float* Wo1 = (const float*)d_in[14]; const float* bo1  = (const float*)d_in[15];
    const float* Wo2 = (const float*)d_in[16]; const float* bo2  = (const float*)d_in[17];
    float* out = (float*)d_out;

    const size_t per = (size_t)BSZ * HH * NSEQ * DHEAD;   // 2,097,152 halfs (4 MB)
    h16* ws = (h16*)d_ws;
    h16 *T1 = ws,           *T2 = ws + per;               // q2 / q1 temporaries
    h16 *Q1 = ws + 2 * per, *Q2 = ws + 3 * per;
    h16 *K1 = ws + 4 * per, *K2 = ws + 5 * per;
    h16 *V1 = ws + 6 * per, *V2 = ws + 7 * per;
    h16 *O1 = T1, *O2 = T2;                               // reuse after Q1/Q2 consume them

    const float scale = 0.125f;                           // 1/sqrt(64)
    dim3 blk(256);
    dim3 gproj(DM / 64, MROWS / 128);                     // (8, 32)

    // q2 = grids@Wq12^T + bq12 ; q1 = regions@Wq12^T + bq12
    gemm_nt_kernel<false,false,false><<<gproj, blk, 0, stream>>>(grids,   nullptr, Wq12, bq12, nullptr, 1.0f, nullptr, T1);
    gemm_nt_kernel<false,false,false><<<gproj, blk, 0, stream>>>(regions, nullptr, Wq12, bq12, nullptr, 1.0f, nullptr, T2);
    // Q1 = scale*(interests@Wq^T + bq + q2) ; Q2 = scale*(... + q1)
    gemm_nt_kernel<false,false,true ><<<gproj, blk, 0, stream>>>(interests, nullptr, Wq, bq, T1, scale, nullptr, Q1);
    gemm_nt_kernel<false,false,true ><<<gproj, blk, 0, stream>>>(interests, nullptr, Wq, bq, T2, scale, nullptr, Q2);
    // K1/K2, V1/V2
    gemm_nt_kernel<false,false,false><<<gproj, blk, 0, stream>>>(regions, nullptr, Wk, bk, nullptr, 1.0f, nullptr, K1);
    gemm_nt_kernel<false,false,false><<<gproj, blk, 0, stream>>>(grids,   nullptr, Wk, bk, nullptr, 1.0f, nullptr, K2);
    gemm_nt_kernel<false,false,false><<<gproj, blk, 0, stream>>>(regions, nullptr, Wv, bv, nullptr, 1.0f, nullptr, V1);
    gemm_nt_kernel<false,false,false><<<gproj, blk, 0, stream>>>(grids,   nullptr, Wv, bv, nullptr, 1.0f, nullptr, V2);

    // fused flash attention
    dim3 gatt(NSEQ / 128, HH, BSZ);                       // (16, 8, 2)
    flash_attn_kernel<<<gatt, blk, 0, stream>>>(Q1, K1, V1, mask_r, attw, O1);
    flash_attn_kernel<<<gatt, blk, 0, stream>>>(Q2, K2, V2, mask_g, attw, O2);

    // output projections -> f32 to d_out
    gemm_nt_kernel<true,true,false><<<gproj, blk, 0, stream>>>(nullptr, O1, Wo1, bo1, nullptr, 1.0f, out, nullptr);
    gemm_nt_kernel<true,true,false><<<gproj, blk, 0, stream>>>(nullptr, O2, Wo2, bo2, nullptr, 1.0f, out + (size_t)MROWS * DM, nullptr);
}
